// HGNN_11536282157341
// MI455X (gfx1250) — matrix operand order repaired and checked by reading, not compiled, not run on hardware
//
#include <hip/hip_runtime.h>
#include <hip/hip_bf16.h>

// ---------------------------------------------------------------------------
// MI455X (gfx1250) HGNN forward.  wave32, WMMA 16x16x32 bf16 (fp32 accum).
// Pipeline is HBM-bandwidth bound (~500MB traffic @ 23.3TB/s); all dense GEMMs
// go through v_wmma_f32_16x16x32_bf16.  Fragment loads are vectorized to
// ds_load_b128 pairs via fragment-contiguous LDS layouts (B stored transposed).
// ---------------------------------------------------------------------------

typedef __attribute__((ext_vector_type(16))) __bf16          v16bf;
typedef __attribute__((ext_vector_type(8)))  float           v8f;
typedef __attribute__((ext_vector_type(8)))  unsigned short  u16x8;
typedef __attribute__((ext_vector_type(4)))  unsigned short  u16x4;
typedef __attribute__((ext_vector_type(4)))  float           f32x4;

union Frag { v16bf v; u16x8 h[2]; unsigned short u[16]; };

__device__ __forceinline__ unsigned short f2bf(float f) {
    unsigned int u = __float_as_uint(f);
    u += 0x7FFFu + ((u >> 16) & 1u);            // round-to-nearest-even
    return (unsigned short)(u >> 16);
}

// problem constants
#define NN   131072
#define EE   524288
#define FF   128
#define DD   256
#define HIDN 1024
#define NEXP 4
#define TSL  128
#define NB   64
#define OUTC 60
#define NCLU 8192
#define CAPC 64
#define NEGV -1000000.0f

// ---------------------------------------------------------------------------
// Generic batched GEMM:  C[z] = act( A[z](MxK) @ B[z/gdiv](KxN) + bias )
// bf16 WMMA, BM=BN=64, BK=32, 128 threads (4 waves).
// LDS:  As row-major 64x32 (A fragment = two contiguous 8-ushort runs),
//       Bs TRANSPOSED 64x32 (col-major) so a B fragment is 16 contiguous
//       ushorts -> two ds_load_b128 per fragment.
// ---------------------------------------------------------------------------
__launch_bounds__(128)
__global__ void gemm_wmma(const float* __restrict__ A, const float* __restrict__ Bw,
                          const float* __restrict__ bias, float* __restrict__ C,
                          int M, int K, int Ncols, int ldc,
                          long long sA, long long sB, long long sC, int gdiv, int act)
{
    __shared__ unsigned short As[64 * 32];
    __shared__ unsigned short Bs[64 * 32];   // Bs[col][k]

    int z = blockIdx.z;
    A  += (long long)z * sA;
    Bw += (long long)(z / gdiv) * sB;
    C  += (long long)z * sC;

    const int m0   = blockIdx.x * 64;
    const int n0   = blockIdx.y * 64;
    const int tid  = threadIdx.x;
    const int wave = tid >> 5;
    const int lane = tid & 31;
    const int half = lane >> 4;
    const int mn   = lane & 15;
    const int r0   = wave * 16;
    const bool fullN = (n0 + 64 <= Ncols);

    v8f acc[4] = {};

    for (int k0 = 0; k0 < K; k0 += 32) {
        // stage A 64x32 (fp32 -> bf16), float4 loads, 8-byte LDS stores
        for (int i = tid; i < 64 * 8; i += 128) {
            int r = i >> 3, c4 = (i & 7) << 2;
            f32x4 v = *(const f32x4*)(A + (size_t)(m0 + r) * K + k0 + c4);
            u16x4 o = { f2bf(v.x), f2bf(v.y), f2bf(v.z), f2bf(v.w) };
            *(u16x4*)(As + r * 32 + c4) = o;
        }
        // stage B 32x64 transposed into Bs[col][k]
        if (fullN) {
            for (int i = tid; i < 32 * 16; i += 128) {
                int r = i >> 4, c4 = (i & 15) << 2;
                f32x4 v = *(const f32x4*)(Bw + (size_t)(k0 + r) * Ncols + n0 + c4);
                Bs[(c4 + 0) * 32 + r] = f2bf(v.x);
                Bs[(c4 + 1) * 32 + r] = f2bf(v.y);
                Bs[(c4 + 2) * 32 + r] = f2bf(v.z);
                Bs[(c4 + 3) * 32 + r] = f2bf(v.w);
            }
        } else {
            for (int i = tid; i < 32 * 64; i += 128) {
                int r = i >> 6, c = i & 63;
                int gn = n0 + c;
                float val = (gn < Ncols) ? Bw[(size_t)(k0 + r) * Ncols + gn] : 0.f;
                Bs[c * 32 + r] = f2bf(val);
            }
        }
        if (k0 + 32 < K)  // hint next A tile into cache (global_prefetch_b8)
            __builtin_prefetch(&A[(size_t)(m0 + (tid >> 1)) * K + k0 + 32], 0, 1);
        __syncthreads();

        // A fragment: row r0+mn, ks [half*16..+15] contiguous
        Frag a;
        const u16x8* arow = (const u16x8*)(As + (r0 + mn) * 32);
        a.h[0] = arow[2 * half];
        a.h[1] = arow[2 * half + 1];
#pragma unroll
        for (int t = 0; t < 4; ++t) {
            // B fragment: column t*16+mn, ks [half*16..+15] contiguous
            Frag b;
            const u16x8* bcol = (const u16x8*)(Bs + (t * 16 + mn) * 32);
            b.h[0] = bcol[2 * half];
            b.h[1] = bcol[2 * half + 1];
            acc[t] = __builtin_amdgcn_wmma_f32_16x16x32_bf16(
                         false, a.v, false, b.v, (short)0, acc[t], false, false);
        }
        __syncthreads();
    }

#pragma unroll
    for (int t = 0; t < 4; ++t) {
        int col = n0 + t * 16 + mn;
        if (col >= Ncols) continue;
        float bv = bias ? bias[col] : 0.f;
#pragma unroll
        for (int r = 0; r < 8; ++r) {
            int row = m0 + r0 + half * 8 + r;               // C: M = half*8 + vgpr
            float v = acc[t][r] + bv;
            if (act == 1) v = v > 0.f ? v : 0.01f * v;      // leaky_relu(0.01)
            C[(size_t)row * ldc + col] = v;
        }
    }
}

// ---------------------------------------------------------------------------
// small utility kernels
// ---------------------------------------------------------------------------
__global__ void fill_f32(float* p, float v, int n) {
    int i = blockIdx.x * blockDim.x + threadIdx.x;
    if (i < n) p[i] = v;
}
__global__ void fill_i32(int* p, int v, int n) {
    int i = blockIdx.x * blockDim.x + threadIdx.x;
    if (i < n) p[i] = v;
}
__global__ void row_bias(float* p, const float* b, int n, int cols) {
    int i = blockIdx.x * blockDim.x + threadIdx.x;
    if (i < n) p[i] = b[i % cols];
}

// as[i] = xp[i,:]·a_src ; ad[i] = xp[i,:]·a_dst   (one wave32 per node)
__global__ void dot_as_ad(const float* __restrict__ xp, const float* __restrict__ a_src,
                          const float* __restrict__ a_dst, float* as_, float* ad_, int n)
{
    int wid  = (blockIdx.x * blockDim.x + threadIdx.x) >> 5;
    int lane = threadIdx.x & 31;
    if (wid >= n) return;
    const float* row = xp + (size_t)wid * FF;
    float s = 0.f, d = 0.f;
    for (int j = lane; j < FF; j += 32) { float v = row[j]; s += v * a_src[j]; d += v * a_dst[j]; }
    for (int o = 16; o; o >>= 1) { s += __shfl_down(s, o, 32); d += __shfl_down(d, o, 32); }
    if (lane == 0) { as_[wid] = s; ad_[wid] = d; }
}

// leaky-relu edge scores (+self loops) and ordered-int atomic segment max
__global__ void edge_score(const int* __restrict__ src, const int* __restrict__ dst,
                           const float* __restrict__ as_, const float* __restrict__ ad_,
                           float* esc, int* emax)
{
    int t = blockIdx.x * blockDim.x + threadIdx.x;
    if (t >= EE + NN) return;
    int s = (t < EE) ? src[t] : (t - EE);
    int d = (t < EE) ? dst[t] : (t - EE);
    float e = as_[s] + ad_[d];
    e = e > 0.f ? e : 0.2f * e;
    esc[t] = e;
    int fi = __float_as_int(e);
    fi = fi >= 0 ? fi : (int)(fi ^ 0x7fffffff);             // order-preserving encode
    atomicMax(emax + d, fi);
}

__global__ void edge_exp(const int* __restrict__ dst, float* esc,
                         const int* __restrict__ emax, float* denom)
{
    int t = blockIdx.x * blockDim.x + threadIdx.x;
    if (t >= EE + NN) return;
    int d = (t < EE) ? dst[t] : (t - EE);
    int fi = emax[d];
    float m = __int_as_float(fi >= 0 ? fi : (int)(fi ^ 0x7fffffff));
    float ex = __expf(esc[t] - m);
    esc[t] = ex;
    atomicAdd(denom + d, ex);
}

// gout[d,:] += alpha * xp[s,:]   (wave per edge)
__global__ void gat_aggregate(const int* __restrict__ src, const int* __restrict__ dst,
                              const float* __restrict__ esc, const float* __restrict__ denom,
                              const float* __restrict__ xp, float* gout)
{
    int wid  = (blockIdx.x * blockDim.x + threadIdx.x) >> 5;
    int lane = threadIdx.x & 31;
    if (wid >= EE + NN) return;
    int s = (wid < EE) ? src[wid] : (wid - EE);
    int d = (wid < EE) ? dst[wid] : (wid - EE);
    float alpha = esc[wid] / denom[d];
    const float* xr = xp + (size_t)s * FF;
    float* orow = gout + (size_t)d * FF;
    for (int j = lane; j < FF; j += 32) atomicAdd(orow + j, xr[j] * alpha);
}

// h2[dst, 128+j] += h2[src, j]   (real edges only)
__global__ void aggr_sum(const int* __restrict__ src, const int* __restrict__ dst, float* h2)
{
    int wid  = (blockIdx.x * blockDim.x + threadIdx.x) >> 5;
    int lane = threadIdx.x & 31;
    if (wid >= EE) return;
    const float* hr = h2 + (size_t)src[wid] * 256;
    float* orow = h2 + (size_t)dst[wid] * 256 + 128;
    for (int j = lane; j < 128; j += 32) atomicAdd(orow + j, hr[j]);
}

// cluster max-pool (cluster = node/16) + column sum-of-squares
__global__ void pool_max(const float* __restrict__ h2, float* pooled, float* colsq)
{
    int i = blockIdx.x * blockDim.x + threadIdx.x;
    if (i >= NCLU * 256) return;
    int c = i >> 8, j = i & 255;
    const float* base = h2 + (size_t)c * 16 * 256 + j;
    float m = -3.4e38f;
    for (int t = 0; t < 16; ++t) m = fmaxf(m, base[(size_t)t * 256]);
    pooled[i] = m;
    atomicAdd(colsq + j, m * m);
}
__global__ void col_norm(float* pooled, const float* colsq)
{
    int i = blockIdx.x * blockDim.x + threadIdx.x;
    if (i < NCLU * 256) pooled[i] /= sqrtf(colsq[i & 255]);
}

// ---------------------------------------------------------------------------
// Masked self-attention: one block per (batch, 16-row q tile).  128 thr / 4 waves.
// scores = q k^T via WMMA (K=256), masked softmax in LDS, out = P v via WMMA.
// ---------------------------------------------------------------------------
__launch_bounds__(128)
__global__ void attn_wmma(const float* __restrict__ Q, const float* __restrict__ Km,
                          const float* __restrict__ V, const int* __restrict__ vl,
                          float* __restrict__ O)
{
    __shared__ float          S[16 * 128];
    __shared__ unsigned short P[16 * 128];

    const int b = blockIdx.y, qt = blockIdx.x;
    const int tid = threadIdx.x, wave = tid >> 5, lane = tid & 31;
    const int half = lane >> 4, mn = lane & 15;
    const int vlen = vl[b];
    const size_t base = (size_t)b * TSL * DD;

    // --- scores: wave handles key tiles {wave, wave+4} ---
    v8f acc[2] = {};
    for (int d0 = 0; d0 < DD; d0 += 32) {
        Frag a;
        const float* qrow = Q + base + (size_t)(qt * 16 + mn) * DD + d0;
        {
            f32x4 q0 = *(const f32x4*)(qrow + half * 16);
            f32x4 q1 = *(const f32x4*)(qrow + half * 16 + 4);
            f32x4 q2 = *(const f32x4*)(qrow + half * 16 + 8);
            f32x4 q3 = *(const f32x4*)(qrow + half * 16 + 12);
            a.u[0] = f2bf(q0.x); a.u[1] = f2bf(q0.y); a.u[2]  = f2bf(q0.z); a.u[3]  = f2bf(q0.w);
            a.u[4] = f2bf(q1.x); a.u[5] = f2bf(q1.y); a.u[6]  = f2bf(q1.z); a.u[7]  = f2bf(q1.w);
            a.u[8] = f2bf(q2.x); a.u[9] = f2bf(q2.y); a.u[10] = f2bf(q2.z); a.u[11] = f2bf(q2.w);
            a.u[12]= f2bf(q3.x); a.u[13]= f2bf(q3.y); a.u[14] = f2bf(q3.z); a.u[15] = f2bf(q3.w);
        }
#pragma unroll
        for (int j = 0; j < 2; ++j) {
            int tc = (wave + 4 * j) * 16;
            Frag bf;
            const float* kc = Km + base + (size_t)(tc + mn) * DD + d0 + half * 16;
#pragma unroll
            for (int q4 = 0; q4 < 4; ++q4) {
                f32x4 kv = *(const f32x4*)(kc + q4 * 4);
                bf.u[q4 * 4 + 0] = f2bf(kv.x); bf.u[q4 * 4 + 1] = f2bf(kv.y);
                bf.u[q4 * 4 + 2] = f2bf(kv.z); bf.u[q4 * 4 + 3] = f2bf(kv.w);
            }
            acc[j] = __builtin_amdgcn_wmma_f32_16x16x32_bf16(
                         false, a.v, false, bf.v, (short)0, acc[j], false, false);
        }
    }
#pragma unroll
    for (int j = 0; j < 2; ++j) {
        int col = (wave + 4 * j) * 16 + mn;
#pragma unroll
        for (int r = 0; r < 8; ++r) {
            int row = half * 8 + r;
            S[row * 128 + col] = (col < vlen) ? acc[j][r] : NEGV;
        }
    }
    __syncthreads();

    // --- softmax (one thread per row) ---
    if (tid < 16) {
        int row = tid;
        float m = -3.4e38f;
        for (int c = 0; c < 128; ++c) m = fmaxf(m, S[row * 128 + c]);
        float sum = 0.f;
        for (int c = 0; c < 128; ++c) sum += __expf(S[row * 128 + c] - m);
        float inv = 1.f / sum;
        for (int c = 0; c < 128; ++c)
            P[row * 128 + c] = f2bf(__expf(S[row * 128 + c] - m) * inv);
    }
    __syncthreads();

    // --- out = P @ V : wave handles d tiles {wave, wave+4, wave+8, wave+12} ---
    v8f ao[4] = {};
    for (int k0 = 0; k0 < 128; k0 += 32) {
        Frag a;
        const u16x8* prow = (const u16x8*)(P + mn * 128 + k0);
        a.h[0] = prow[2 * half];
        a.h[1] = prow[2 * half + 1];
#pragma unroll
        for (int j = 0; j < 4; ++j) {
            int dc = (wave + 4 * j) * 16;
            Frag bf;
            const float* vr = V + base + (size_t)(k0 + half * 16) * DD + dc + mn;
#pragma unroll
            for (int e = 0; e < 16; ++e) bf.u[e] = f2bf(vr[(size_t)e * DD]);
            ao[j] = __builtin_amdgcn_wmma_f32_16x16x32_bf16(
                        false, a.v, false, bf.v, (short)0, ao[j], false, false);
        }
    }
#pragma unroll
    for (int j = 0; j < 4; ++j) {
        int dc = (wave + 4 * j) * 16 + mn;
#pragma unroll
        for (int r = 0; r < 8; ++r)
            O[base + (size_t)(qt * 16 + half * 8 + r) * DD + dc] = ao[j][r];
    }
}

// ---------------------------------------------------------------------------
// Top-2 gating (eval mode, capacity=64).  One wave per batch: lanes compute
// per-token logits/top2, then lane 0 runs the order-dependent capacity cumsum.
// ---------------------------------------------------------------------------
__global__ void gating_kernel(const float* __restrict__ ao, const float* __restrict__ wg,
                              int* te1, int* tc1, int* te2, int* tc2, float* tw1, float* tw2)
{
    __shared__ int   si1[TSL], si2[TSL], skeep[TSL], sc1[TSL];
    __shared__ float sg1[TSL], sg2[TSL];
    const int b = blockIdx.x, lane = threadIdx.x;

    for (int n = lane; n < TSL; n += 32) {
        const float* row = ao + ((size_t)b * TSL + n) * DD;
        float l[NEXP] = {};
        for (int d = 0; d < DD; ++d) {
            float x = row[d];
            const float* w = wg + d * NEXP;
            l[0] += x * w[0]; l[1] += x * w[1]; l[2] += x * w[2]; l[3] += x * w[3];
        }
        float mx = fmaxf(fmaxf(l[0], l[1]), fmaxf(l[2], l[3]));
        float p[NEXP], sum = 0.f;
        for (int e = 0; e < NEXP; ++e) { p[e] = __expf(l[e] - mx); sum += p[e]; }
        for (int e = 0; e < NEXP; ++e) p[e] /= sum;
        int i1 = 0; float g1 = p[0];
        for (int e = 1; e < NEXP; ++e) if (p[e] > g1) { g1 = p[e]; i1 = e; }
        int i2 = -1; float g2 = -1.f;
        for (int e = 0; e < NEXP; ++e) if (e != i1 && p[e] > g2) { g2 = p[e]; i2 = e; }
        float den = g1 + g2 + 1e-9f;
        float g1n = g1 / den, g2n = g2 / den;
        unsigned int h = (unsigned)(b * TSL + n) * 2654435761u + 42u;
        h ^= h >> 13; h *= 2246822519u; h ^= h >> 16;
        float prob = (h >> 8) * (1.0f / 16777216.0f);
        si1[n] = i1; si2[n] = i2; sg1[n] = g1n; sg2[n] = g2n;
        skeep[n] = (prob < (g2n / 0.2f)) ? 1 : 0;
    }
    __syncthreads();

    if (lane == 0) {
        int cnt[NEXP] = {};
        for (int n = 0; n < TSL; ++n) { int e = si1[n]; int p = cnt[e]++; sc1[n] = (p < CAPC) ? p : -1; }
        int m1[NEXP], cnt2[NEXP] = {};
        for (int e = 0; e < NEXP; ++e) m1[e] = cnt[e] < CAPC ? cnt[e] : CAPC;
        for (int n = 0; n < TSL; ++n) {
            int tok = b * TSL + n;
            te1[tok] = si1[n]; tc1[tok] = sc1[n];
            tw1[tok] = (sc1[n] >= 0) ? sg1[n] : 0.f;
            int c2 = -1; float w2v = 0.f;
            if (skeep[n]) {
                int e = si2[n];
                int p = m1[e] + cnt2[e]++;
                if (p < CAPC) { c2 = p; w2v = sg2[n]; }
            }
            te2[tok] = si2[n]; tc2[tok] = c2; tw2[tok] = w2v;
        }
    }
}

// scatter tokens into expert buffers  ein[e,b,c,:] = out[b,n,:]
__global__ void ein_scatter(const float* __restrict__ ao, const int* te1, const int* tc1,
                            const int* te2, const int* tc2, float* ein)
{
    int i = blockIdx.x * blockDim.x + threadIdx.x;
    if (i >= NB * TSL * DD) return;
    int tok = i >> 8, d = i & 255, b = tok >> 7;
    float v = ao[i];
    if (tc1[tok] >= 0) ein[(((size_t)te1[tok] * NB + b) * CAPC + tc1[tok]) * DD + d] = v;
    if (tc2[tok] >= 0) ein[(((size_t)te2[tok] * NB + b) * CAPC + tc2[tok]) * DD + d] = v;
}

// combine expert outputs
__global__ void moe_combine(const float* __restrict__ eout, const int* te1, const int* tc1,
                            const int* te2, const int* tc2, const float* tw1, const float* tw2,
                            float* moe)
{
    int i = blockIdx.x * blockDim.x + threadIdx.x;
    if (i >= NB * TSL * DD) return;
    int tok = i >> 8, d = i & 255, b = tok >> 7;
    float r = 0.f;
    if (tc1[tok] >= 0) r += tw1[tok] * eout[(((size_t)te1[tok] * NB + b) * CAPC + tc1[tok]) * DD + d];
    if (tc2[tok] >= 0) r += tw2[tok] * eout[(((size_t)te2[tok] * NB + b) * CAPC + tc2[tok]) * DD + d];
    moe[i] = r;
}

// LayerNorm(256) + ReLU, one wave per row
__global__ void ln_relu(const float* __restrict__ X, const float* __restrict__ g,
                        const float* __restrict__ beta, float* __restrict__ Y, int rows)
{
    int wid  = (blockIdx.x * blockDim.x + threadIdx.x) >> 5;
    int lane = threadIdx.x & 31;
    if (wid >= rows) return;
    const float* x = X + (size_t)wid * DD;
    float s = 0.f, s2 = 0.f;
    for (int j = lane; j < DD; j += 32) { float v = x[j]; s += v; s2 += v * v; }
    for (int o = 16; o; o >>= 1) { s += __shfl_down(s, o, 32); s2 += __shfl_down(s2, o, 32); }
    s = __shfl(s, 0, 32); s2 = __shfl(s2, 0, 32);
    float mu = s / (float)DD, var = s2 / (float)DD - mu * mu;
    float inv = rsqrtf(var + 1e-5f);
    float* y = Y + (size_t)wid * DD;
    for (int j = lane; j < DD; j += 32) {
        float v = (x[j] - mu) * inv * g[j] + beta[j];
        y[j] = v > 0.f ? v : 0.f;
    }
}

// ---------------------------------------------------------------------------
static inline void gemm(const float* A, const float* B, const float* bias, float* C,
                        int M, int K, int Nc, int ldc, int act,
                        long long sA, long long sB, long long sC, int gdiv, int nz,
                        hipStream_t st)
{
    dim3 g(M / 64, (Nc + 63) / 64, nz);
    gemm_wmma<<<g, dim3(128), 0, st>>>(A, B, bias, C, M, K, Nc, ldc, sA, sB, sC, gdiv, act);
}

extern "C" void kernel_launch(void* const* d_in, const int* in_sizes, int n_in,
                              void* d_out, int out_size, void* d_ws, size_t ws_size,
                              hipStream_t stream)
{
    (void)in_sizes; (void)n_in; (void)out_size; (void)ws_size;

    const float* x        = (const float*)d_in[0];
    const int*   eidx     = (const int*)d_in[1];
    const int*   src      = eidx;
    const int*   dst      = eidx + EE;
    const int*   vlen     = (const int*)d_in[3];
    const float* gat_W    = (const float*)d_in[5];
    const float* gat_asrc = (const float*)d_in[6];
    const float* gat_adst = (const float*)d_in[7];
    const float* gat_bias = (const float*)d_in[8];
    const float* lin_W    = (const float*)d_in[9];
    const float* lin_b    = (const float*)d_in[10];
    const float* qW = (const float*)d_in[11]; const float* qb = (const float*)d_in[12];
    const float* kW = (const float*)d_in[13]; const float* kb = (const float*)d_in[14];
    const float* vW = (const float*)d_in[15]; const float* vb = (const float*)d_in[16];
    const float* wg = (const float*)d_in[17];
    const float* w1 = (const float*)d_in[18];
    const float* w2 = (const float*)d_in[19];
    const float* mW1 = (const float*)d_in[20]; const float* mb1 = (const float*)d_in[21];
    const float* lng = (const float*)d_in[22]; const float* lnb = (const float*)d_in[23];
    const float* mW2 = (const float*)d_in[24]; const float* mb2 = (const float*)d_in[25];
    float* out = (float*)d_out;

    // ---- workspace layout (bytes) ----
    char* ws = (char*)d_ws;
    float* XP   = (float*)(ws);                          // 64MB  (later: POOLED)
    float* GOUT = (float*)(ws + (size_t)67108864);       // 64MB  (later: Q,K,V,AO)
    float* H2   = (float*)(ws + (size_t)134217728);      // 128MB (later: EIN..HH2)
    char*  SM   = ws + (size_t)268435456;                // small region
    float* AS    = (float*)(SM);
    float* AD    = (float*)(SM + 524288);
    float* ESC   = (float*)(SM + 1048576);               // (E+N) floats
    int*   EMAXI = (int*)  (SM + 3670016);
    float* DENOM = (float*)(SM + 4194304);
    float* COLSQ = (float*)(SM + 4718592);
    int*   TE1   = (int*)  (SM + 4720640);
    int*   TC1   = (int*)  (SM + 4753408);
    int*   TE2   = (int*)  (SM + 4786176);
    int*   TC2   = (int*)  (SM + 4818944);
    float* TW1   = (float*)(SM + 4851712);
    float* TW2   = (float*)(SM + 4884480);
    // aliases (lifetimes disjoint)
    float* POOLED = XP;
    float* Qb = GOUT, *Kb = GOUT + 2097152, *Vb = GOUT + 4194304, *AO = GOUT + 6291456;
    float* EIN  = H2;
    float* HID  = H2 + 4194304;
    float* EOUT = HID + 16777216;
    float* MOE  = EOUT + 4194304;
    float* HH   = MOE + 2097152;
    float* HH2  = HH + 2097152;

    const int T = 256;
    #define CDIV(a,b) (((a)+(b)-1)/(b))

    // ---- 1. GAT ----
    fill_f32<<<CDIV(NN*256, T), T, 0, stream>>>(H2, 0.f, NN * 256);
    fill_f32<<<CDIV(NN, T), T, 0, stream>>>(DENOM, 0.f, NN);
    fill_f32<<<1, T, 0, stream>>>(COLSQ, 0.f, 256);
    fill_i32<<<CDIV(NN, T), T, 0, stream>>>(EMAXI, (int)0x80000000, NN);
    row_bias<<<CDIV(NN*FF, T), T, 0, stream>>>(GOUT, gat_bias, NN * FF, FF);

    gemm(x, gat_W, nullptr, XP, NN, FF, FF, FF, 0, 0, 0, 0, 1, 1, stream);     // xp = x@W
    dot_as_ad<<<CDIV(NN*32, T), T, 0, stream>>>(XP, gat_asrc, gat_adst, AS, AD, NN);
    edge_score<<<CDIV(EE+NN, T), T, 0, stream>>>(src, dst, AS, AD, ESC, EMAXI);
    edge_exp<<<CDIV(EE+NN, T), T, 0, stream>>>(dst, ESC, EMAXI, DENOM);
    gat_aggregate<<<CDIV((EE+NN)*32, T), T, 0, stream>>>(src, dst, ESC, DENOM, XP, GOUT);

    // ---- 2. linear + neighbor aggregate + concat ----
    gemm(GOUT, lin_W, lin_b, H2, NN, FF, FF, 256, 0, 0, 0, 0, 1, 1, stream);   // h -> H2[:,:128]
    aggr_sum<<<CDIV(EE*32, T), T, 0, stream>>>(src, dst, H2);                  // -> H2[:,128:]

    // ---- 3. pool + column L2 norm ----
    pool_max<<<CDIV(NCLU*256, T), T, 0, stream>>>(H2, POOLED, COLSQ);
    col_norm<<<CDIV(NCLU*256, T), T, 0, stream>>>(POOLED, COLSQ);

    // ---- 4. self-attention ----
    gemm(POOLED, qW, qb, Qb, NCLU, DD, DD, DD, 0, 0, 0, 0, 1, 1, stream);
    gemm(POOLED, kW, kb, Kb, NCLU, DD, DD, DD, 0, 0, 0, 0, 1, 1, stream);
    gemm(POOLED, vW, vb, Vb, NCLU, DD, DD, DD, 0, 0, 0, 0, 1, 1, stream);
    attn_wmma<<<dim3(TSL/16, NB), dim3(128), 0, stream>>>(Qb, Kb, Vb, vlen, AO);

    // ---- 5. MoE ----
    gating_kernel<<<NB, 32, 0, stream>>>(AO, wg, TE1, TC1, TE2, TC2, TW1, TW2);
    fill_f32<<<CDIV(NEXP*NB*CAPC*DD, T), T, 0, stream>>>(EIN, 0.f, NEXP*NB*CAPC*DD);
    ein_scatter<<<CDIV(NB*TSL*DD, T), T, 0, stream>>>(AO, TE1, TC1, TE2, TC2, EIN);
    gemm(EIN, w1, nullptr, HID, CAPC, DD, HIDN, HIDN, 1,
         (long long)CAPC*DD, (long long)DD*HIDN, (long long)CAPC*HIDN, NB, NEXP*NB, stream);
    gemm(HID, w2, nullptr, EOUT, CAPC, HIDN, DD, DD, 0,
         (long long)CAPC*HIDN, (long long)HIDN*DD, (long long)CAPC*DD, NB, NEXP*NB, stream);
    moe_combine<<<CDIV(NB*TSL*DD, T), T, 0, stream>>>(EOUT, TE1, TC1, TE2, TC2, TW1, TW2, MOE);

    // ---- 6. TrajPredMLP ----
    gemm(MOE, mW1, mb1, HH, NB*TSL, DD, DD, DD, 0, 0, 0, 0, 1, 1, stream);
    ln_relu<<<CDIV(NB*TSL*32, T), T, 0, stream>>>(HH, lng, lnb, HH2, NB*TSL);
    gemm(HH2, mW2, mb2, out, NB*TSL, DD, OUTC, OUTC, 0, 0, 0, 0, 1, 1, stream);
}
